// S4Block1D_54511724921192
// MI455X (gfx1250) — matrix operand order repaired and checked by reading, not compiled, not run on hardware
//
#include <hip/hip_runtime.h>
#include <hip/hip_bf16.h>
#include <math.h>

#define Bq   8
#define Hq   512
#define Lq   2048
#define N2q  32
#define BLq  (Bq * Lq)       // 16384 columns
#define KPAD 128
#define LKP  (Lq + KPAD)     // padded kernel row stride

typedef __attribute__((ext_vector_type(16))) __bf16 bf16x16;
typedef __attribute__((ext_vector_type(8)))  float  f32x8;

union FragAB { bf16x16 v; unsigned u[8]; };

// round-to-nearest-even f32 -> bf16 (raw bits)
__device__ __forceinline__ unsigned short f2bf(float f) {
  unsigned u = __float_as_uint(f);
  u += 0x7FFFu + ((u >> 16) & 1u);
  return (unsigned short)(u >> 16);
}

// A-fragment of the Toeplitz kernel matrix from a per-lane pointer:
// element pair for vgpr j sits at constant offsets -k and -k-1 (k in
// {0,2,4,6,16,18,20,22}); the g*8 lane term is folded into the pointer.
__device__ __forceinline__ void load_fragA(FragAB& F, const unsigned short* __restrict__ p) {
#pragma unroll
  for (int j = 0; j < 8; ++j) {
    const int k = ((j & 3) << 1) + ((j >> 2) << 4);
    F.u[j] = (unsigned)p[-k] | ((unsigned)p[-k - 1] << 16);
  }
}

// ---------------- K1: per-(h,n) discretization ----------------
__global__ void k1_prep(const float* __restrict__ log_dt,
                        const float* __restrict__ C_re, const float* __restrict__ C_im,
                        const float* __restrict__ logAr, const float* __restrict__ A_im,
                        float* __restrict__ aRe, float* __restrict__ aIm,
                        float* __restrict__ cR,  float* __restrict__ cI) {
  int t = blockIdx.x * blockDim.x + threadIdx.x;
  if (t >= Hq * N2q) return;
  int h = t / N2q;
  float dt  = expf(log_dt[h]);
  float ar  = -expf(logAr[t]);
  float ai  = A_im[t];
  float dAr = ar * dt, dAi = ai * dt;
  float ex  = expf(dAr);
  float nR  = ex * cosf(dAi) - 1.0f;     // exp(dtA) - 1
  float nI  = ex * sinf(dAi);
  float inv = 1.0f / (ar * ar + ai * ai);
  float qR  = (nR * ar + nI * ai) * inv; // (exp(dtA)-1)/A
  float qI  = (nI * ar - nR * ai) * inv;
  float cr = C_re[t], ci = C_im[t];
  aRe[t] = dAr;  aIm[t] = dAi;
  cR[t] = 2.0f * (cr * qR - ci * qI);    // fold factor 2 of 2*Re(...)
  cI[t] = 2.0f * (cr * qI + ci * qR);
}

// ---------------- K2: materialize zero-padded SSM kernel K[h,l] in bf16 ----------------
__global__ __launch_bounds__(256) void k2_genK(
    const float* __restrict__ aRe, const float* __restrict__ aIm,
    const float* __restrict__ cR,  const float* __restrict__ cI,
    unsigned short* __restrict__ Kpad) {
  __shared__ float sAr[N2q], sAi[N2q], sWr[N2q], sWi[N2q], sCr[N2q], sCi[N2q];
  int h = blockIdx.x;
  int t = threadIdx.x;
  if (t < N2q) {
    float ar = aRe[h * N2q + t], ai = aIm[h * N2q + t];
    sAr[t] = ar; sAi[t] = ai;
    float e = expf(ar);
    sWr[t] = e * cosf(ai);
    sWi[t] = e * sinf(ai);
    sCr[t] = cR[h * N2q + t];
    sCi[t] = cI[h * N2q + t];
  }
  // zero the 128-entry negative-d pad (threads 0..15 write 8 each)
  if (t < KPAD / 8) {
#pragma unroll
    for (int i = 0; i < 8; ++i) Kpad[(size_t)h * LKP + t * 8 + i] = 0;
  }
  __syncthreads();
  float l0 = (float)(t * 8);
  float acc[8] = {0.f,0.f,0.f,0.f,0.f,0.f,0.f,0.f};
  for (int n = 0; n < N2q; ++n) {
    float e  = expf(sAr[n] * l0);
    float pR = e * cosf(sAi[n] * l0);
    float pI = e * sinf(sAi[n] * l0);
    float wR = sWr[n], wI = sWi[n], cv = sCr[n], dv = sCi[n];
#pragma unroll
    for (int i = 0; i < 8; ++i) {
      acc[i] += cv * pR - dv * pI;       // Re(C~ * w^l)
      float tR = pR * wR - pI * wI;
      pI = pR * wI + pI * wR;
      pR = tR;
    }
  }
  size_t base = (size_t)h * LKP + KPAD + t * 8;
#pragma unroll
  for (int i = 0; i < 8; ++i) Kpad[base + i] = f2bf(acc[i]);
}

// ---------------- K0: f32 -> bf16 conversion ----------------
__global__ void k0_cvt(const float* __restrict__ in, unsigned short* __restrict__ o, int n) {
  int t = blockIdx.x * blockDim.x + threadIdx.x;
  if (t < n) o[t] = f2bf(in[t]);
}

// ---------------- K3: causal Toeplitz conv via WMMA + D-skip + GELU ----------------
// One wave per (h, 64-wide l group): 4 l-tiles share B fragments; the Toeplitz
// A window slides (A_i(j0+32) == A_{i-2}(j0)) so only 2 new A fragments load
// per iteration. All hot-loop addresses are moving pointers + const offsets.
__global__ __launch_bounds__(32) void k3_conv(
    const unsigned short* __restrict__ Kpad, const unsigned short* __restrict__ xb,
    const float* __restrict__ x, const float* __restrict__ Dv,
    unsigned short* __restrict__ y1s) {
  const int gt = blockIdx.x, h = blockIdx.y;
  const int lane = threadIdx.x;
  const int l0 = gt << 6;       // group covers l in [l0, l0+64)
  const int m  = lane & 15;     // M row within tile (output l)
  const int g  = lane >> 4;     // half-wave select
  const unsigned short* Kp = Kpad + (size_t)h * LKP + KPAD;  // valid from index -128
  const unsigned short* pA = Kp + (l0 + m) - (g << 3);       // per-lane A pointer
  const unsigned short* pB = xb + ((size_t)((lane & 7) * Hq + h)) * Lq + (g << 3);
  f32x8 acc0 = {}, acc1 = {}, acc2 = {}, acc3 = {};
  FragAB A0, A1, A2, A3;
  load_fragA(A0, pA);
  load_fragA(A1, pA + 16);
  load_fragA(A2, pA + 32);
  load_fragA(A3, pA + 48);
  const int iters = (gt << 1) + 2;       // ceil((l0+64)/32)
#pragma unroll 2
  for (int it = 0; it < iters; ++it) {
    FragAB Bf;
#pragma unroll
    for (int j = 0; j < 8; ++j)          // 2 runs of 4 dwords -> 2x b128
      Bf.u[j] = *(const unsigned*)(pB + (((j & 3) << 1) + ((j >> 2) << 4)));
    acc0 = __builtin_amdgcn_wmma_f32_16x16x32_bf16(false, A0.v, false, Bf.v,
                                                   (short)0, acc0, false, false);
    acc1 = __builtin_amdgcn_wmma_f32_16x16x32_bf16(false, A1.v, false, Bf.v,
                                                   (short)0, acc1, false, false);
    acc2 = __builtin_amdgcn_wmma_f32_16x16x32_bf16(false, A2.v, false, Bf.v,
                                                   (short)0, acc2, false, false);
    acc3 = __builtin_amdgcn_wmma_f32_16x16x32_bf16(false, A3.v, false, Bf.v,
                                                   (short)0, acc3, false, false);
    // slide the Toeplitz window: A_i' = A_{i-2}; load 2 new fragments
    A3 = A1; A2 = A0;
    pA -= 32;
    pB += 32;
    load_fragA(A0, pA);
    load_fragA(A1, pA + 16);
  }
  const int nb = lane & 15;              // N column = batch index (valid < 8)
  if (nb < Bq) {
    float d = Dv[h];
    const int hp = h >> 1, hb = h & 1;
    f32x8 accs[4] = {acc0, acc1, acc2, acc3};
#pragma unroll
    for (int i = 0; i < 4; ++i) {
#pragma unroll
      for (int r = 0; r < 8; ++r) {
        int l = l0 + (i << 4) + r + (g << 3);
        float v = accs[i][r] + d * x[((size_t)(nb * Hq + h)) * Lq + l];
        v = 0.5f * v * (1.0f + erff(v * 0.70710678118654752f));   // exact GELU
        size_t c = (size_t)nb * Lq + l;
        y1s[((size_t)hp * BLq + c) * 2 + hb] = f2bf(v);  // pair-interleaved layout
      }
    }
  }
}

// ---------------- K4: output linear (WMMA) + bias + GLU ----------------
// One wave per (32-row o block, 64-col c block): 2 o-pair tiles x 4 c tiles.
// A reused 4x across columns, B reused 4x across rows; pointer+const-offset
// addressing in the k loop.
__global__ __launch_bounds__(32) void k4_gemm_glu(
    const unsigned short* __restrict__ Wb, const unsigned* __restrict__ y1p,
    const float* __restrict__ bout, float* __restrict__ y2) {
  const int mo = blockIdx.x;             // o rows [mo*32, mo*32+32) (a-half)
  const int nt = blockIdx.y;             // columns [nt*64, nt*64+64)
  const int lane = threadIdx.x;
  const int m = lane & 15, g = lane >> 4;
  const int cn = (nt << 6) + m;
  const unsigned short* pa0 = Wb + (size_t)((mo << 5) + m) * Hq + (g << 3);
  const unsigned short* pa1 = pa0 + (size_t)16 * Hq;
  const unsigned short* pg0 = pa0 + (size_t)Hq * Hq;   // +512 rows
  const unsigned short* pg1 = pg0 + (size_t)16 * Hq;
  const unsigned*       pb  = y1p + (size_t)(g << 2) * BLq + cn;
  f32x8 accA[2][4] = {}, accG[2][4] = {};
  for (int kt = 0; kt < Hq / 32; ++kt) {
    FragAB Aa[2], Ag[2], Bf[4];
#pragma unroll
    for (int j = 0; j < 8; ++j) {
      const int k = ((j & 3) << 1) + ((j >> 2) << 4);   // const element offsets
      Aa[0].u[j] = *(const unsigned*)(pa0 + k);
      Aa[1].u[j] = *(const unsigned*)(pa1 + k);
      Ag[0].u[j] = *(const unsigned*)(pg0 + k);
      Ag[1].u[j] = *(const unsigned*)(pg1 + k);
    }
#pragma unroll
    for (int t4 = 0; t4 < 4; ++t4) {
#pragma unroll
      for (int j = 0; j < 8; ++j)
        Bf[t4].u[j] = pb[(size_t)((j & 3) + ((j >> 2) << 3)) * BLq + (t4 << 4)];  // const offsets
    }
#pragma unroll
    for (int t2 = 0; t2 < 2; ++t2) {
#pragma unroll
      for (int t4 = 0; t4 < 4; ++t4) {
        accA[t2][t4] = __builtin_amdgcn_wmma_f32_16x16x32_bf16(
            false, Aa[t2].v, false, Bf[t4].v, (short)0, accA[t2][t4], false, false);
        accG[t2][t4] = __builtin_amdgcn_wmma_f32_16x16x32_bf16(
            false, Ag[t2].v, false, Bf[t4].v, (short)0, accG[t2][t4], false, false);
      }
    }
    pa0 += 32; pa1 += 32; pg0 += 32; pg1 += 32;   // advance K by 32
    pb  += (size_t)16 * BLq;                      // 32 rows = 16 dword-pairs
  }
#pragma unroll
  for (int t2 = 0; t2 < 2; ++t2) {
#pragma unroll
    for (int t4 = 0; t4 < 4; ++t4) {
#pragma unroll
      for (int r = 0; r < 8; ++r) {
        int o = (mo << 5) + (t2 << 4) + r + (g << 3);
        int c = cn + (t4 << 4);
        float za = accA[t2][t4][r] + bout[o];
        float zg = accG[t2][t4][r] + bout[o + Hq];
        y2[(size_t)o * BLq + c] = za * (1.0f / (1.0f + expf(-zg)));  // GLU
      }
    }
  }
}

// ---------------- K5: channel LayerNorm + ReLU + mask ----------------
__global__ __launch_bounds__(256) void k5_ln(
    const float* __restrict__ y2, const unsigned char* __restrict__ msk,
    const float* __restrict__ gamma, const float* __restrict__ beta,
    float* __restrict__ out) {
  int wave = (int)((blockIdx.x * blockDim.x + threadIdx.x) >> 5);
  int lane = threadIdx.x & 31;
  if (wave >= BLq) return;
  int b = wave / Lq, l = wave - b * Lq;
  float vals[16], s = 0.f, s2 = 0.f;
#pragma unroll
  for (int i = 0; i < 16; ++i) {
    int h = (i << 5) + lane;
    float v = y2[(size_t)h * BLq + wave];
    vals[i] = v; s += v; s2 += v * v;
  }
#pragma unroll
  for (int off = 16; off; off >>= 1) {
    s  += __shfl_xor(s,  off, 32);
    s2 += __shfl_xor(s2, off, 32);
  }
  float mu   = s  * (1.0f / Hq);
  float var  = s2 * (1.0f / Hq) - mu * mu;
  float rstd = rsqrtf(var + 1e-5f);
  bool masked = (msk[wave] != 0);
#pragma unroll
  for (int i = 0; i < 16; ++i) {
    int h = (i << 5) + lane;
    float o = (vals[i] - mu) * rstd * gamma[h] + beta[h];
    o = fmaxf(o, 0.0f);
    if (masked) o = 0.0f;
    out[((size_t)(b * Hq + h)) * Lq + l] = o;
  }
}

// ---------------- host-side launch ----------------
extern "C" void kernel_launch(void* const* d_in, const int* in_sizes, int n_in,
                              void* d_out, int out_size, void* d_ws, size_t ws_size,
                              hipStream_t stream) {
  (void)in_sizes; (void)n_in; (void)out_size; (void)ws_size;
  const float*         x      = (const float*)d_in[0];
  const unsigned char* x_mask = (const unsigned char*)d_in[1];   // (B,1,L) bool
  const float*         log_dt = (const float*)d_in[2];
  const float*         C_re   = (const float*)d_in[3];
  const float*         C_im   = (const float*)d_in[4];
  const float*         logAr  = (const float*)d_in[5];
  const float*         A_im   = (const float*)d_in[6];
  const float*         Dv     = (const float*)d_in[7];
  const float*         W_out  = (const float*)d_in[8];
  const float*         b_out  = (const float*)d_in[9];
  const float*         gamma  = (const float*)d_in[10];
  const float*         beta   = (const float*)d_in[11];
  float* out = (float*)d_out;

  char* ws = (char*)d_ws;
  auto alloc = [&](size_t bytes) -> char* {
    char* p = ws; ws += (bytes + 255) & ~(size_t)255; return p;
  };
  float*          aRe  = (float*)alloc((size_t)Hq * N2q * 4);
  float*          aIm  = (float*)alloc((size_t)Hq * N2q * 4);
  float*          cR   = (float*)alloc((size_t)Hq * N2q * 4);
  float*          cI   = (float*)alloc((size_t)Hq * N2q * 4);
  unsigned short* Kpad = (unsigned short*)alloc((size_t)Hq * LKP * 2);      // ~2.2 MB
  unsigned short* xb   = (unsigned short*)alloc((size_t)Bq * Hq * Lq * 2);  // 16 MB
  unsigned short* Wb   = (unsigned short*)alloc((size_t)2 * Hq * Hq * 2);   // 1 MB
  unsigned short* y1s  = (unsigned short*)alloc((size_t)Hq * BLq * 2);      // 16 MB
  float*          y2   = (float*)alloc((size_t)Hq * BLq * 4);               // 33.5 MB

  k1_prep<<<(Hq * N2q + 255) / 256, 256, 0, stream>>>(log_dt, C_re, C_im, logAr, A_im,
                                                      aRe, aIm, cR, cI);
  k2_genK<<<Hq, 256, 0, stream>>>(aRe, aIm, cR, cI, Kpad);
  {
    int n = Bq * Hq * Lq;
    k0_cvt<<<(n + 255) / 256, 256, 0, stream>>>(x, xb, n);
    int nw = 2 * Hq * Hq;
    k0_cvt<<<(nw + 255) / 256, 256, 0, stream>>>(W_out, Wb, nw);
  }
  k3_conv<<<dim3(Lq / 64, Hq), 32, 0, stream>>>(Kpad, xb, x, Dv, y1s);
  k4_gemm_glu<<<dim3(Hq / 32, BLq / 64), 32, 0, stream>>>(Wb, (const unsigned*)y1s,
                                                          b_out, y2);
  k5_ln<<<(BLq * 32 + 255) / 256, 256, 0, stream>>>(y2, x_mask, gamma, beta, out);
}
